// GRUCell_61581241090199
// MI455X (gfx1250) — matrix operand order repaired and checked by reading, not compiled, not run on hardware
//
#include <hip/hip_runtime.h>
#include <hip/hip_bf16.h>

// GRU on gfx1250: persistent recurrence kernel, bf16 WMMA (f32 accumulate),
// async global->LDS double-buffered staging of x_t tiles, 2 barriers/step.
// B=128, T=2048, H=128.

#define BDIM 128
#define TDIM 2048
#define HDIM 128

typedef __attribute__((ext_vector_type(16))) __bf16 v16bf;
typedef __attribute__((ext_vector_type(8)))  __bf16 v8bf;
typedef __attribute__((ext_vector_type(8)))  float  v8f;

#if defined(__has_builtin)
#if __has_builtin(__builtin_amdgcn_global_load_async_to_lds_b128) && \
    __has_builtin(__builtin_amdgcn_s_wait_asynccnt)
#define HAVE_ASYNC_LDS 1
#endif
#if __has_builtin(__builtin_nontemporal_store)
#define HAVE_NT_STORE 1
#endif
#endif
#ifndef HAVE_ASYNC_LDS
#define HAVE_ASYNC_LDS 0
#endif
#ifndef HAVE_NT_STORE
#define HAVE_NT_STORE 0
#endif

#if HAVE_ASYNC_LDS
// Exact parameter types per hipcc diagnostic:
//   arg0: __attribute__((__vector_size__(4*sizeof(int)))) int __device__ *
//   arg1: LDS-side pointer of the same vector type
//   arg2: constant integer offset, arg3: constant cpol
typedef __attribute__((__vector_size__(4 * sizeof(int)))) int async_b128_t;
typedef __attribute__((address_space(1))) async_b128_t* gbl_b128p;
typedef __attribute__((address_space(3))) async_b128_t* lds_b128p;
#endif

static __device__ __forceinline__ v16bf pack16(v8bf lo, v8bf hi) {
  return __builtin_shufflevector(lo, hi, 0, 1, 2, 3, 4, 5, 6, 7,
                                         8, 9, 10, 11, 12, 13, 14, 15);
}

static __device__ __forceinline__ v8bf cvt8(float4 a, float4 b) {
  v8bf c;
  c[0] = (__bf16)a.x; c[1] = (__bf16)a.y; c[2] = (__bf16)a.z; c[3] = (__bf16)a.w;
  c[4] = (__bf16)b.x; c[5] = (__bf16)b.y; c[6] = (__bf16)b.z; c[7] = (__bf16)b.w;
  return c;
}

// A-fragment (16-bit A 16x32 layout): lane M=lane&15, half=lane>>4;
//   elems 0..7  -> K = kb + half*8 + e
//   elems 8..15 -> K = kb + 16 + half*8 + (e-8)
static __device__ __forceinline__ v16bf afrag_bf16(const __bf16* Arow,
                                                   int half, int kb) {
  v8bf lo = *(const v8bf*)(Arow + kb + half * 8);
  v8bf hi = *(const v8bf*)(Arow + kb + 16 + half * 8);
  return pack16(lo, hi);
}
static __device__ __forceinline__ v16bf afrag_f32(const float* Arow,
                                                  int half, int kb) {
  const float4* p0 = (const float4*)(Arow + kb + half * 8);
  const float4* p1 = (const float4*)(Arow + kb + 16 + half * 8);
  return pack16(cvt8(p0[0], p0[1]), cvt8(p1[0], p1[1]));
}

// D += A(frags) x B, where B[k][n] = W[i0+n][k] (row-major bf16 weight rows).
// B-fragment (16-bit B 32x16 layout): lane N=lane&15, half=lane>>4;
//   elems 0..7  -> K = kb + half*8 + e
//   elems 8..15 -> K = kb + 16 + half*8 + (e-8)
static __device__ __forceinline__ v8f gate_gemm(const v16bf* af,
                                                const __bf16* Brow,
                                                int half, v8f acc) {
#pragma unroll
  for (int kk = 0; kk < 4; ++kk) {
    const int kb = kk * 32;
    v8bf blo = *(const v8bf*)(Brow + kb + half * 8);
    v8bf bhi = *(const v8bf*)(Brow + kb + 16 + half * 8);
    acc = __builtin_amdgcn_wmma_f32_16x16x32_bf16(
        /*neg_a=*/false, af[kk], /*neg_b=*/false, pack16(blo, bhi),
        /*c_mod=*/(short)0, acc, /*reuse_a=*/false, /*reuse_b=*/false);
  }
  return acc;
}

// Async copy of one 16-byte chunk; OFF must be a compile-time constant
// (the builtin requires a constant integer; INST_OFFSET applies to both the
// LDS and global addresses).
template <int OFF>
static __device__ __forceinline__ void async_b128(const void* gsrc,
                                                  void* ldst) {
#if HAVE_ASYNC_LDS
  __builtin_amdgcn_global_load_async_to_lds_b128(
      (gbl_b128p)gsrc, (lds_b128p)ldst, OFF, 0);
#else
  *(float4*)((char*)ldst + OFF) = *(const float4*)((const char*)gsrc + OFF);
#endif
}
static __device__ __forceinline__ void stage_wait() {
#if HAVE_ASYNC_LDS
  __builtin_amdgcn_s_wait_asynccnt(0);
#endif
}

static __device__ __forceinline__ void store_nt(float v, float* p) {
#if HAVE_NT_STORE
  __builtin_nontemporal_store(v, p);
#else
  *p = v;
#endif
}

static __device__ __forceinline__ float sigmoid_f(float x) {
  return 1.0f / (1.0f + __expf(-x));
}
static __device__ __forceinline__ float tanh_f(float x) {
  float ax = fabsf(x);
  float e  = __expf(-2.0f * ax);
  float t  = (1.0f - e) / (1.0f + e);
  return copysignf(t, x);
}

// Convert the six [H][H] f32 weight matrices to a packed bf16 copy in ws.
// Order: Wz, Wr, Wh, Uz, Ur, Uh (row-major, same layout as source).
__global__ __launch_bounds__(256) void gru_cvt_weights(
    const float* __restrict__ wz, const float* __restrict__ wr,
    const float* __restrict__ wh, const float* __restrict__ uz,
    const float* __restrict__ ur, const float* __restrict__ uh,
    __bf16* __restrict__ dst) {
  int i = blockIdx.x * 256 + threadIdx.x;  // 0 .. 6*H*H-1
  int m = i >> 14;                         // which matrix (H*H == 16384)
  int o = i & 16383;
  const float* s = (m == 0) ? wz : (m == 1) ? wr : (m == 2) ? wh
                 : (m == 3) ? uz : (m == 4) ? ur : uh;
  dst[i] = (__bf16)s[o];
}

// Whole-tensor f32 -> bf16 conversion of x (memory-bound, parallel pass):
// 134 MB read + 67 MB write ~= 9 us at 23.3 TB/s.
__global__ __launch_bounds__(256) void gru_cvt_x(const float* __restrict__ src,
                                                 __bf16* __restrict__ dst) {
  long i = ((long)blockIdx.x * 256 + threadIdx.x) * 8;
  const float4* s = (const float4*)(src + i);
  *(v8bf*)(dst + i) = cvt8(s[0], s[1]);
}

// Persistent GRU scan. Grid: B/16 blocks of 256 threads (8 waves).
// Block owns batch rows [b0, b0+16). Wave w owns hidden columns
// [16*w, 16*w+16). h tile lives in the wave's WMMA accumulator registers;
// loop-invariant weight B-operands are promoted to high VGPRs (VGPR-MSB).
// XBF16: x was pre-converted to bf16 in ws -> async-stage bf16 tiles, no
// per-step conversion on the serial path.
template <bool XBF16>
__global__ __launch_bounds__(256) void gru_persistent(
    const float* __restrict__ xf32, const __bf16* __restrict__ xb16,
    const int* __restrict__ lengths,
    const float* __restrict__ bz, const float* __restrict__ br,
    const float* __restrict__ bh, const __bf16* __restrict__ wbf,
    float* __restrict__ out) {
  // Double-buffered x_t tiles: f32 mode uses 8KB/buf, bf16 mode first 4KB/buf.
  __shared__ float  xstage[2][16 * HDIM];
  __shared__ __bf16 hbf[16 * HDIM];  // h_{t-1} (bf16, full K for A operands)
  __shared__ __bf16 rbf[16 * HDIM];  // r .* h (bf16)

  const int tid  = threadIdx.x;
  const int lane = tid & 31;
  const int wave = tid >> 5;   // 0..7
  const int i0   = wave * 16;  // N-tile base within H
  const int N    = lane & 15;
  const int half = lane >> 4;
  const int b0   = blockIdx.x * 16;

  const __bf16* WzB = wbf + 0 * HDIM * HDIM + (i0 + N) * HDIM;
  const __bf16* WrB = wbf + 1 * HDIM * HDIM + (i0 + N) * HDIM;
  const __bf16* WhB = wbf + 2 * HDIM * HDIM + (i0 + N) * HDIM;
  const __bf16* UzB = wbf + 3 * HDIM * HDIM + (i0 + N) * HDIM;
  const __bf16* UrB = wbf + 4 * HDIM * HDIM + (i0 + N) * HDIM;
  const __bf16* UhB = wbf + 5 * HDIM * HDIM + (i0 + N) * HDIM;

  // Loop-invariant bias splats (accumulator initializers).
  v8f bz8, br8, bh8;
  {
    const float bzv = bz[i0 + N], brv = br[i0 + N], bhv = bh[i0 + N];
#pragma unroll
    for (int e = 0; e < 8; ++e) { bz8[e] = bzv; br8[e] = brv; bh8[e] = bhv; }
  }

  // Per-lane accumulator element e -> (m = e + 8*half, n = N).
  int    lenv[8];
  float* orow[8];
#pragma unroll
  for (int e = 0; e < 8; ++e) {
    int m   = e + 8 * half;
    lenv[e] = lengths[b0 + m];
    orow[e] = out + ((long)(b0 + m) * TDIM) * HDIM + i0 + N;
  }

  // Per-thread staging slot: 8 consecutive elements of the [16][128] tile.
  const int fi      = tid * 8;
  const int m_stage = fi >> 7;
  const int k_stage = fi & (HDIM - 1);
  const float*  xrow_f = xf32 + ((long)(b0 + m_stage) * TDIM) * HDIM + k_stage;
  const __bf16* xrow_b =
      XBF16 ? (xb16 + ((long)(b0 + m_stage) * TDIM) * HDIM + k_stage) : nullptr;

  v8f hreg;
#pragma unroll
  for (int e = 0; e < 8; ++e) hreg[e] = 0.0f;

#pragma unroll
  for (int j = tid; j < 16 * HDIM; j += 256) hbf[j] = (__bf16)0.0f;

  // Prologue: stage x tile for t=0 into buffer 0.
  if (XBF16) {
    async_b128<0>(xrow_b, (__bf16*)&xstage[0][0] + fi);
  } else {
    async_b128<0>(xrow_f, &xstage[0][fi]);
    async_b128<16>(xrow_f, &xstage[0][fi]);
  }
  stage_wait();
  __syncthreads();

  const __bf16* Ah = hbf + (lane & 15) * HDIM;
  const __bf16* Ar = rbf + (lane & 15) * HDIM;

  for (int t = 0; t < TDIM; ++t) {
    const int cur = t & 1;
    const int nxt = cur ^ 1;

    // Prefetch x[t+1]; the async DMA lands while this step's WMMAs run.
    {
      const long tn = (t + 1 < TDIM) ? (t + 1) : t;
      if (XBF16) {
        async_b128<0>(xrow_b + tn * HDIM, (__bf16*)&xstage[nxt][0] + fi);
      } else {
        async_b128<0>(xrow_f + tn * HDIM, &xstage[nxt][fi]);
        async_b128<16>(xrow_f + tn * HDIM, &xstage[nxt][fi]);
      }
    }

    // Build A fragments once; reuse across gates.
    v16bf ax[4], ahf[4];
#pragma unroll
    for (int kk = 0; kk < 4; ++kk) {
      if (XBF16) {
        const __bf16* AxRow = (const __bf16*)&xstage[cur][0] + (lane & 15) * HDIM;
        ax[kk] = afrag_bf16(AxRow, half, kk * 32);
      } else {
        const float* AxRow = &xstage[cur][0] + (lane & 15) * HDIM;
        ax[kk] = afrag_f32(AxRow, half, kk * 32);
      }
      ahf[kk] = afrag_bf16(Ah, half, kk * 32);
    }

    // z and r pre-activations: x_t @ W^T + h @ U^T + b
    v8f az = bz8, ar = br8;
    az = gate_gemm(ax, WzB, half, az);
    az = gate_gemm(ahf, UzB, half, az);
    ar = gate_gemm(ax, WrB, half, ar);
    ar = gate_gemm(ahf, UrB, half, ar);

    float zg[8];
#pragma unroll
    for (int e = 0; e < 8; ++e) {
      zg[e]    = sigmoid_f(az[e]);
      float rg = sigmoid_f(ar[e]);
      float rh = rg * hreg[e];  // h co-located with r in the D layout
      rbf[(e + 8 * half) * HDIM + i0 + N] = (__bf16)rh;
    }
    __syncthreads();  // (A) r.*h staged; all xstage/hbf reads are done

    // candidate: tanh(x_t @ Wh^T + (r.*h) @ Uh^T + bh)
    v8f ah = bh8;
    {
      v16bf arf[4];
#pragma unroll
      for (int kk = 0; kk < 4; ++kk) arf[kk] = afrag_bf16(Ar, half, kk * 32);
      ah = gate_gemm(ax, WhB, half, ah);
      ah = gate_gemm(arf, UhB, half, ah);
    }

#pragma unroll
    for (int e = 0; e < 8; ++e) {
      float hc = tanh_f(ah[e]);
      float hn = (1.0f - zg[e]) * hreg[e] + zg[e] * hc;
      hreg[e]  = hn;  // f32 state stays in registers
      hbf[(e + 8 * half) * HDIM + i0 + N] = (__bf16)hn;  // next step's A
      store_nt((t < lenv[e]) ? hn : 0.0f, orow[e] + (long)t * HDIM);
    }

    stage_wait();     // x[t+1] landed in xstage[nxt] (ASYNCcnt -> 0)
    __syncthreads();  // (B) h_t and x[t+1] visible to all waves
  }
}

extern "C" void kernel_launch(void* const* d_in, const int* in_sizes, int n_in,
                              void* d_out, int out_size, void* d_ws,
                              size_t ws_size, hipStream_t stream) {
  (void)in_sizes; (void)n_in; (void)out_size;
  const float* x    = (const float*)d_in[0];
  const int*   lens = (const int*)d_in[1];
  const float* Wz   = (const float*)d_in[2];
  const float* Uz   = (const float*)d_in[3];
  const float* bz   = (const float*)d_in[4];
  const float* Wr   = (const float*)d_in[5];
  const float* Ur   = (const float*)d_in[6];
  const float* br   = (const float*)d_in[7];
  const float* Wh   = (const float*)d_in[8];
  const float* Uh   = (const float*)d_in[9];
  const float* bh   = (const float*)d_in[10];
  float* out = (float*)d_out;

  const size_t need_w = (size_t)6 * HDIM * HDIM * sizeof(__bf16);  // 196608 B
  const size_t need_x =
      (size_t)BDIM * TDIM * HDIM * sizeof(__bf16);                 // 67.1 MB

  __bf16* wbf = (__bf16*)d_ws;
  __bf16* xws = (__bf16*)((char*)d_ws + need_w);

  gru_cvt_weights<<<(6 * HDIM * HDIM) / 256, 256, 0, stream>>>(
      Wz, Wr, Wh, Uz, Ur, Uh, wbf);

  if (ws_size >= need_w + need_x) {
    gru_cvt_x<<<(BDIM * TDIM * HDIM / 8) / 256, 256, 0, stream>>>(x, xws);
    gru_persistent<true><<<BDIM / 16, 256, 0, stream>>>(
        x, xws, lens, bz, br, bh, wbf, out);
  } else {
    gru_persistent<false><<<BDIM / 16, 256, 0, stream>>>(
        x, nullptr, lens, bz, br, bh, wbf, out);
  }
}
